// HeadPredictor_1228360647012
// MI455X (gfx1250) — matrix operand order repaired and checked by reading, not compiled, not run on hardware
//
#include <hip/hip_runtime.h>

// ---------------------------------------------------------------------------
// MI455X (gfx1250) implementation of the FOMM-style keypoint head.
//   Kernel 1: 7x7 conv as 49 shifted K=64 GEMMs via v_wmma_f32_16x16x32_bf16
//   Kernel 2: fused softmax + moments + 2x2 eig + gaussian
// Logits are staged in the gaussian region of d_out (same element count),
// so no workspace is required. Bias bc is skipped: softmax over h*w is
// invariant to a per-(n,c) constant shift, so bc cancels from every output.
// ---------------------------------------------------------------------------

typedef __attribute__((ext_vector_type(16))) __bf16 v16bf;
typedef __attribute__((ext_vector_type(8)))  float  v8f;

struct alignas(16) Q16 { unsigned int x, y, z, w; };
union FragBF { Q16 q[2]; v16bf v; };

#define HW        64
#define CIN       64
#define COUT      10
#define NCH       16      // cout padded to WMMA N
#define ROWS      8       // output rows per workgroup
#define PATCH_H   14      // ROWS + 6
#define PATCH_W   70      // 64 + 6
#define HOMO_N    23040                    // 256*10*9
#define IMG_N     10485760                 // 256*10*64*64
#define HEAT_OFF  HOMO_N
#define GAUSS_OFF (HOMO_N + IMG_N)

__device__ __forceinline__ unsigned short f32_to_bf16(float f) {
  unsigned int u = __float_as_uint(f);
  unsigned int r = (u + 0x7FFFu + ((u >> 16) & 1u)) >> 16;  // RNE
  return (unsigned short)r;
}

// ---------------------------------------------------------------------------
// Kernel 1: conv via WMMA.  grid = 256 * 8 blocks, 256 threads (8 waves).
// Block (n, rb) computes logits for rows [rb*8, rb*8+8) of image n,
// all 10 (padded 16) output channels.  Each wave owns one output row
// (4 M-tiles of 16 pixels), accumulating over 49 taps x 2 K-steps.
// ---------------------------------------------------------------------------
__global__ __launch_bounds__(256) void conv_wmma_kernel(
    const float* __restrict__ x,    // [256][64][64][64]
    const float* __restrict__ Wc,   // [10][64][7][7]
    float*       __restrict__ logits /* [256][10][64][64] */) {

  __shared__ __align__(16) unsigned short sX[PATCH_H][PATCH_W][CIN]; // bf16
  __shared__ __align__(16) unsigned short sB[NCH][CIN];              // bf16

  const int bx    = blockIdx.x;
  const int n     = bx >> 3;
  const int rb    = bx & 7;
  const int ybase = rb * ROWS;
  const int tid   = threadIdx.x;

  // ---- stage input patch (zero padded) into LDS, channel-minor bf16 ----
  const int PATCH_ELEMS = PATCH_H * PATCH_W * CIN;
  for (int idx = tid; idx < PATCH_ELEMS; idx += 256) {
    int px = idx % PATCH_W;
    int r  = idx / PATCH_W;
    int c  = r & (CIN - 1);
    int py = r >> 6;
    int iy = ybase - 3 + py;
    int ix = px - 3;
    float v = 0.0f;
    if ((unsigned)iy < 64u && (unsigned)ix < 64u)
      v = x[(((size_t)n * CIN + c) * HW + iy) * HW + ix];
    sX[py][px][c] = f32_to_bf16(v);
  }

  const int wave = tid >> 5;
  const int lane = tid & 31;
  const int mrow = lane & 15;   // A row / B column (= output channel)
  const int half = lane >> 4;   // K-half selector
  const int oy   = ybase + wave;

  v8f acc[4];
  const v8f vzero = {0.f, 0.f, 0.f, 0.f, 0.f, 0.f, 0.f, 0.f};
#pragma unroll
  for (int t = 0; t < 4; ++t) acc[t] = vzero;

  for (int tap = 0; tap < 49; ++tap) {
    const int ky = tap / 7;
    const int kx = tap - ky * 7;

    __syncthreads();   // previous tap's compute done before sB overwrite
    // ---- stage this tap's weights: sB[nch][c] = Wc[nch][c][ky][kx] ----
    for (int idx = tid; idx < NCH * CIN; idx += 256) {
      int c   = idx & (CIN - 1);
      int nch = idx >> 6;
      float v = (nch < COUT) ? Wc[((size_t)nch * CIN + c) * 49 + tap] : 0.0f;
      sB[nch][c] = f32_to_bf16(v);
    }
    __syncthreads();

    const int py = wave + ky;
#pragma unroll
    for (int ksub = 0; ksub < 2; ++ksub) {
      const int kk = ksub * 32;

      // B fragment (32x16): lane -> column mrow; K = half*16 + j, contiguous
      FragBF bf;
      {
        const Q16* bp = (const Q16*)&sB[mrow][kk + half * 16];
        bf.q[0] = bp[0];
        bf.q[1] = bp[1];
      }

      const int ac0 = kk + half * 16; // A: K runs {ac0..+15}, contiguous
#pragma unroll
      for (int t = 0; t < 4; ++t) {
        const int px = t * 16 + mrow + kx;
        FragBF af;
        af.q[0] = *(const Q16*)&sX[py][px][ac0];
        af.q[1] = *(const Q16*)&sX[py][px][ac0 + 8];
        acc[t] = __builtin_amdgcn_wmma_f32_16x16x32_bf16(
            false, af.v, false, bf.v, (short)0, acc[t], false, false);
      }
    }
  }

  // ---- epilogue: C/D layout -> lane holds channel mrow, M = half*8 + r ----
  const int ch = mrow;
  if (ch < COUT) {
    float* lrow = &logits[(((size_t)n * COUT + ch) * HW + oy) * HW];
#pragma unroll
    for (int t = 0; t < 4; ++t)
#pragma unroll
      for (int r = 0; r < 8; ++r)
        lrow[t * 16 + half * 8 + r] = acc[t][r];
  }
}

// ---------------------------------------------------------------------------
// Kernel 2: per (n,c) image: softmax -> heatmap, moments -> mean/covar,
// eig -> homo_affine, covar^-1 -> gaussian (overwrites the staged logits).
// grid = 2560 blocks, 256 threads, 16 pixels/thread.
// ---------------------------------------------------------------------------
__global__ __launch_bounds__(256) void postproc_kernel(float* out) {
  float* homo  = out;
  float* heat  = out + HEAT_OFF;
  float* gauss = out + GAUSS_OFF;   // currently holds logits

  __shared__ float red[256];
  __shared__ float sred[5][256];
  __shared__ float par[6];

  const int    b    = blockIdx.x;           // n*10 + c
  const int    tid  = threadIdx.x;
  const size_t base = (size_t)b * 4096;

  const float GS = 2.0f / 63.0f;
  const int   h  = tid >> 2;                // (tid*16) >> 6
  const int   w0 = (tid * 16) & 63;
  const float gy = (float)h * GS - 1.0f;

  // ---- load 16 logits ----
  float l[16];
#pragma unroll
  for (int q = 0; q < 4; ++q) {
    float4 v = *(const float4*)&gauss[base + (size_t)tid * 16 + q * 4];
    l[q * 4 + 0] = v.x; l[q * 4 + 1] = v.y; l[q * 4 + 2] = v.z; l[q * 4 + 3] = v.w;
  }

  // ---- max reduce ----
  float m = -3.402823466e38f;
#pragma unroll
  for (int k = 0; k < 16; ++k) m = fmaxf(m, l[k]);
  red[tid] = m;
  __syncthreads();
  for (int s = 128; s > 0; s >>= 1) {
    if (tid < s) red[tid] = fmaxf(red[tid], red[tid + s]);
    __syncthreads();
  }
  const float M = red[0];
  __syncthreads();

  // ---- exp + sum reduce ----
  float ssum = 0.0f;
#pragma unroll
  for (int k = 0; k < 16; ++k) { l[k] = __expf(l[k] - M); ssum += l[k]; }
  red[tid] = ssum;
  __syncthreads();
  for (int s = 128; s > 0; s >>= 1) {
    if (tid < s) red[tid] += red[tid + s];
    __syncthreads();
  }
  const float rinv = 1.0f / red[0];

  // ---- heatmap + moment partials ----
  float sx = 0.f, sy = 0.f, sxx = 0.f, sxy = 0.f, syy = 0.f;
#pragma unroll
  for (int q = 0; q < 4; ++q) {
    float4 pv;
    float p0 = l[q * 4 + 0] * rinv, p1 = l[q * 4 + 1] * rinv;
    float p2 = l[q * 4 + 2] * rinv, p3 = l[q * 4 + 3] * rinv;
    pv.x = p0; pv.y = p1; pv.z = p2; pv.w = p3;
    *(float4*)&heat[base + (size_t)tid * 16 + q * 4] = pv;
    float pp[4] = {p0, p1, p2, p3};
#pragma unroll
    for (int j = 0; j < 4; ++j) {
      float gx = (float)(w0 + q * 4 + j) * GS - 1.0f;
      float p  = pp[j];
      sx  += p * gx;       sy  += p * gy;
      sxx += p * gx * gx;  sxy += p * gx * gy;  syy += p * gy * gy;
    }
  }
  sred[0][tid] = sx;  sred[1][tid] = sy;
  sred[2][tid] = sxx; sred[3][tid] = sxy; sred[4][tid] = syy;
  __syncthreads();
  for (int s = 128; s > 0; s >>= 1) {
    if (tid < s) {
#pragma unroll
      for (int j = 0; j < 5; ++j) sred[j][tid] += sred[j][tid + s];
    }
    __syncthreads();
  }

  // ---- thread 0: covar, eig, affine ----
  if (tid == 0) {
    float mx  = sred[0][0], my = sred[1][0];
    float c00 = sred[2][0] - mx * mx;
    float c01 = sred[3][0] - mx * my;
    float c11 = sred[4][0] - my * my;

    float half_tr = 0.5f * (c00 + c11);
    float dd      = 0.25f * (c00 - c11) * (c00 - c11) + c01 * c01;
    float disc    = sqrtf(fmaxf(dd, 0.0f));
    float l1 = half_tr + disc, l2 = half_tr - disc;
    bool  off = fabsf(c01) > 1e-12f;
    float vx  = off ? c01        : (c00 >= c11 ? 1.0f : 0.0f);
    float vy  = off ? (l1 - c00) : (c00 >= c11 ? 0.0f : 1.0f);
    float nrm = sqrtf(vx * vx + vy * vy);
    vx /= nrm; vy /= nrm;
    float s1 = sqrtf(fmaxf(l1, 0.0f)), s2 = sqrtf(fmaxf(l2, 0.0f));

    float* h9 = homo + (size_t)b * 9;
    h9[0] = vx * s1;  h9[1] = -vy * s2;  h9[2] = mx;
    h9[3] = vy * s1;  h9[4] =  vx * s2;  h9[5] = my;
    h9[6] = 0.0f;     h9[7] = 0.0f;      h9[8] = 1.0f;

    par[0] = mx;  par[1] = my;
    par[2] = c00; par[3] = c01; par[4] = c11;
    par[5] = c00 * c11 - c01 * c01;   // det
  }
  __syncthreads();

  // ---- gaussian: exp(-0.5 * dev^T covar^-1 dev), overwrite logits ----
  const float mx = par[0], my = par[1];
  const float a = par[2], bb = par[3], d = par[4];
  const float idet = 1.0f / par[5];
  const float dy = gy - my;
#pragma unroll
  for (int q = 0; q < 4; ++q) {
    float4 gv;
    float g4[4];
#pragma unroll
    for (int j = 0; j < 4; ++j) {
      float gx = (float)(w0 + q * 4 + j) * GS - 1.0f;
      float dx = gx - mx;
      float ue = (d * dx * dx - 2.0f * bb * dx * dy + a * dy * dy) * idet;
      g4[j] = __expf(-0.5f * ue);
    }
    gv.x = g4[0]; gv.y = g4[1]; gv.z = g4[2]; gv.w = g4[3];
    *(float4*)&gauss[base + (size_t)tid * 16 + q * 4] = gv;
  }
}

// ---------------------------------------------------------------------------
extern "C" void kernel_launch(void* const* d_in, const int* in_sizes, int n_in,
                              void* d_out, int out_size, void* d_ws, size_t ws_size,
                              hipStream_t stream) {
  const float* x  = (const float*)d_in[0];
  const float* Wc = (const float*)d_in[1];
  // d_in[2] = bc: per-channel bias, cancelled by softmax shift-invariance.
  float* out = (float*)d_out;

  float* logits = out + GAUSS_OFF;   // stage logits in the gaussian region
  conv_wmma_kernel<<<dim3(256 * 8), dim3(256), 0, stream>>>(x, Wc, logits);
  postproc_kernel<<<dim3(2560), dim3(256), 0, stream>>>(out);
}
